// Seq2SeqWithAttention_50586124812747
// MI455X (gfx1250) — matrix operand order repaired and checked by reading, not compiled, not run on hardware
//
#include <hip/hip_runtime.h>

// ---------------------------------------------------------------------------
// Problem constants
// ---------------------------------------------------------------------------
#define BATCH 32
#define SSRC  128
#define TDEC  128        // S_TGT - 1
#define HDIM  512
#define CVOC  32000
#define K2    1024       // 2*H
#define LDSTRIDE 520     // padded LDS row stride in bf16 elems (1040B: 16B aligned, bank-spread)

typedef unsigned short u16;
typedef unsigned int   u32;
typedef __attribute__((ext_vector_type(16))) __bf16 bf16x16;
typedef __attribute__((ext_vector_type(8)))  float  f32x8;

union Frag {
    bf16x16 v;
    uint4   q[2];
    u16     s[16];
};

__device__ __forceinline__ float bf2f(u16 h) {
    return __uint_as_float(((u32)h) << 16);
}
__device__ __forceinline__ u16 f2bf(float f) {
    u32 u = __float_as_uint(f);
    u32 r = (u + 0x7FFFu + ((u >> 16) & 1u)) >> 16;   // round-to-nearest-even
    return (u16)r;
}

// Opaque zero offset: blocks LICM from hoisting loop-invariant fragment loads
// (which previously spilled to scratch) WITHOUT destroying address-space
// inference -- pointer arithmetic stays rooted at the kernarg pointer, so the
// loads remain global_load_b128 (saddr+imm) rather than flat_load_b128
// (flat would also tick DScnt and couple against the LDS pipeline).
__device__ __forceinline__ size_t opaque_zero() {
    size_t z = 0;
    asm volatile("" : "+s"(z));
    return z;
}

// A fragment (16x32 bf16, M x K), rows contiguous in k.
// lane L: M = L%16 ; half = L/16 ; elems [0..7]=K(half*8+0..7), [8..15]=K(16+half*8+0..7)
__device__ __forceinline__ Frag load_frag_A_g(const u16* __restrict__ base,
                                              int row0, int rowstride, int k0, int lane) {
    Frag f;
    int row  = row0 + (lane & 15);
    int half = lane >> 4;
    const u16* p = base + (size_t)row * rowstride + k0 + half * 8;
    f.q[0] = *(const uint4*)(p);
    f.q[1] = *(const uint4*)(p + 16);
    return f;
}

__device__ __forceinline__ Frag load_frag_A_lds(const u16* base, int row0, int k0, int lane) {
    Frag f;
    int row  = row0 + (lane & 15);
    int half = lane >> 4;
    const u16* p = base + row * LDSTRIDE + k0 + half * 8;
    f.q[0] = *(const uint4*)(p);
    f.q[1] = *(const uint4*)(p + 16);
    return f;
}

// B fragment (32x16 bf16, K x N), B[k][n] = W[n][k], W row-major [N][Kstride].
// lane L: N = L%16 ; khalf = L/16 ; elems [0..7]=K(k0+khalf*8+0..7), [8..15]=K(k0+16+khalf*8+0..7)
__device__ __forceinline__ Frag load_frag_B_g(const u16* __restrict__ W,
                                              int n0, int Kstride, int k0, int lane) {
    Frag f;
    int n     = n0 + (lane & 15);
    int khalf = lane >> 4;
    const u16* p = W + (size_t)n * Kstride + k0 + khalf * 8;
    f.q[0] = *(const uint4*)(p);
    f.q[1] = *(const uint4*)(p + 16);
    return f;
}

__device__ __forceinline__ f32x8 wmma_bf16(const Frag& a, const Frag& b, f32x8 c) {
    return __builtin_amdgcn_wmma_f32_16x16x32_bf16(false, a.v, false, b.v,
                                                   (short)0, c, false, false);
}

// ---------------------------------------------------------------------------
// Prep kernels
// ---------------------------------------------------------------------------
__global__ void cvt_bf16_kernel(const float* __restrict__ src, u16* __restrict__ dst, int n) {
    int i = blockIdx.x * blockDim.x + threadIdx.x;
    if (i < n) dst[i] = f2bf(src[i]);
}

__global__ void addbias_kernel(const float* __restrict__ a, const float* __restrict__ b,
                               float* __restrict__ d, int n) {
    int i = blockIdx.x * blockDim.x + threadIdx.x;
    if (i < n) d[i] = a[i] + b[i];
}

// out[t*BATCH + b][0..H) = bf16(embed[idx[b*rowstride + t]][:])   grid = TDEC*BATCH
__global__ void gather_kernel(const int* __restrict__ idx, int rowstride,
                              const float* __restrict__ embed, u16* __restrict__ out) {
    int tb = blockIdx.x;
    int t  = tb >> 5;
    int b  = tb & 31;
    int row = idx[b * rowstride + t];
    const float* e = embed + (size_t)row * HDIM;
    u16* o = out + (size_t)tb * HDIM;
    for (int i = threadIdx.x; i < HDIM; i += blockDim.x) o[i] = f2bf(e[i]);
}

// ---------------------------------------------------------------------------
// Encoder RNN: persistent single workgroup, 32 waves, h lives in LDS.
// x_seq: [S][B][H] bf16 ; Wih/Whh: [H][H] bf16 row-major ; bias = bih+bhh f32
// enc_out: [S][B][H] bf16
// ---------------------------------------------------------------------------
__global__ __launch_bounds__(1024) void encoder_kernel(
    const u16* __restrict__ x_seq, const u16* __restrict__ Wih,
    const u16* __restrict__ Whh, const float* __restrict__ bias,
    u16* __restrict__ enc_out) {
    __shared__ u16 lds_h[BATCH * LDSTRIDE];

    int lane = threadIdx.x & 31;
    int wave = threadIdx.x >> 5;          // 0..31, owns N-tile `wave`
    for (int i = threadIdx.x; i < BATCH * LDSTRIDE; i += 1024) lds_h[i] = 0;
    __syncthreads();

    int n0   = wave * 16;
    float bv = bias[n0 + (lane & 15)];
    int half = lane >> 4;
    int nn   = n0 + (lane & 15);

    for (int t = 0; t < SSRC; ++t) {
        // keep weight streams inside the loop (L2-resident), opaque offset
        // defeats LICM while preserving GLOBAL address-space inference
        size_t z = opaque_zero();
        const u16* wih = Wih + z;
        const u16* whh = Whh + z;

        f32x8 acc0, acc1;
        #pragma unroll
        for (int r = 0; r < 8; ++r) { acc0[r] = bv; acc1[r] = bv; }

        const u16* xrow = x_seq + (size_t)t * BATCH * HDIM;
        for (int k0 = 0; k0 < HDIM; k0 += 32) {
            Frag b  = load_frag_B_g(wih, n0, HDIM, k0, lane);
            Frag a0 = load_frag_A_g(xrow, 0, HDIM, k0, lane);
            Frag a1 = load_frag_A_g(xrow, 16, HDIM, k0, lane);
            acc0 = wmma_bf16(a0, b, acc0);
            acc1 = wmma_bf16(a1, b, acc1);
        }
        for (int k0 = 0; k0 < HDIM; k0 += 32) {
            Frag b  = load_frag_B_g(whh, n0, HDIM, k0, lane);
            Frag a0 = load_frag_A_lds(lds_h, 0, k0, lane);
            Frag a1 = load_frag_A_lds(lds_h, 16, k0, lane);
            acc0 = wmma_bf16(a0, b, acc0);
            acc1 = wmma_bf16(a1, b, acc1);
        }
        __syncthreads();   // all reads of lds_h done before overwrite

        u16* orow = enc_out + (size_t)t * BATCH * HDIM;
        #pragma unroll
        for (int r = 0; r < 8; ++r) {
            int m0 = r + 8 * half;
            u16 b0 = f2bf(tanhf(acc0[r]));
            u16 b1 = f2bf(tanhf(acc1[r]));
            lds_h[m0 * LDSTRIDE + nn]        = b0;
            lds_h[(m0 + 16) * LDSTRIDE + nn] = b1;
            orow[(size_t)m0 * HDIM + nn]        = b0;
            orow[(size_t)(m0 + 16) * HDIM + nn] = b1;
        }
        __syncthreads();
    }
}

// ---------------------------------------------------------------------------
// enc_t = enc_bsh @ attn_W^T + attn_b   (M=4096, N=512, K=512), wave-per-tile
// A = enc_out flattened [4096][512]; out enc_t [s*B+b][512] bf16
// grid (32 ntiles, 32), block 256 (8 waves -> 8 mtiles)
// ---------------------------------------------------------------------------
__global__ __launch_bounds__(256) void attn_gemm_kernel(
    const u16* __restrict__ A, const u16* __restrict__ W,
    const float* __restrict__ bias, u16* __restrict__ out) {
    int wave = threadIdx.x >> 5, lane = threadIdx.x & 31;
    int mtile = blockIdx.y * 8 + wave;
    int ntile = blockIdx.x;
    int m0 = mtile * 16, n0 = ntile * 16;

    f32x8 acc;
    float bv = bias[n0 + (lane & 15)];
    #pragma unroll
    for (int r = 0; r < 8; ++r) acc[r] = bv;

    for (int k0 = 0; k0 < HDIM; k0 += 32) {
        Frag a = load_frag_A_g(A, m0, HDIM, k0, lane);
        Frag b = load_frag_B_g(W, n0, HDIM, k0, lane);
        acc = wmma_bf16(a, b, acc);
    }
    int half = lane >> 4, nn = n0 + (lane & 15);
    #pragma unroll
    for (int r = 0; r < 8; ++r) {
        int m = m0 + r + 8 * half;
        out[(size_t)m * HDIM + nn] = f2bf(acc[r]);
    }
}

// ---------------------------------------------------------------------------
// Decoder RNN + attention: persistent single workgroup (32 waves).
// x_seq: dec_emb [T][B][H] bf16 ; enc_out/enc_t: [S][B][H] bf16
// model_hidden: [T*B][2H] bf16 ; out_att: f32 att_weight[b][s][t]
// ---------------------------------------------------------------------------
__global__ __launch_bounds__(1024) void decoder_kernel(
    const u16* __restrict__ x_seq, const u16* __restrict__ Wih,
    const u16* __restrict__ Whh, const float* __restrict__ bias,
    const u16* __restrict__ enc_out, const u16* __restrict__ enc_t,
    u16* __restrict__ model_hidden, float* __restrict__ out_att) {
    __shared__ u16   lds_h[BATCH * LDSTRIDE];
    __shared__ float lds_sc[BATCH * SSRC];

    int lane = threadIdx.x & 31;
    int wave = threadIdx.x >> 5;

    // h0 = enc_outputs[-1]
    for (int i = threadIdx.x; i < BATCH * HDIM; i += 1024) {
        int b = i >> 9, k = i & 511;
        lds_h[b * LDSTRIDE + k] = enc_out[(size_t)(SSRC - 1) * BATCH * HDIM + i];
    }
    __syncthreads();

    int n0   = wave * 16;
    float bv = bias[n0 + (lane & 15)];
    int half = lane >> 4;
    int nn   = n0 + (lane & 15);

    for (int t = 0; t < TDEC; ++t) {
        // opaque per-step offset: these bases are t-invariant and would
        // otherwise be hoistable into massive spilled live ranges
        size_t z = opaque_zero();
        const u16* wih = Wih + z;
        const u16* whh = Whh + z;
        const u16* etp = enc_t + z;
        const u16* eop = enc_out + z;

        // ---- h update (WMMA) ----
        f32x8 acc0, acc1;
        #pragma unroll
        for (int r = 0; r < 8; ++r) { acc0[r] = bv; acc1[r] = bv; }

        const u16* xrow = x_seq + (size_t)t * BATCH * HDIM;
        for (int k0 = 0; k0 < HDIM; k0 += 32) {
            Frag b  = load_frag_B_g(wih, n0, HDIM, k0, lane);
            Frag a0 = load_frag_A_g(xrow, 0, HDIM, k0, lane);
            Frag a1 = load_frag_A_g(xrow, 16, HDIM, k0, lane);
            acc0 = wmma_bf16(a0, b, acc0);
            acc1 = wmma_bf16(a1, b, acc1);
        }
        for (int k0 = 0; k0 < HDIM; k0 += 32) {
            Frag b  = load_frag_B_g(whh, n0, HDIM, k0, lane);
            Frag a0 = load_frag_A_lds(lds_h, 0, k0, lane);
            Frag a1 = load_frag_A_lds(lds_h, 16, k0, lane);
            acc0 = wmma_bf16(a0, b, acc0);
            acc1 = wmma_bf16(a1, b, acc1);
        }
        __syncthreads();
        #pragma unroll
        for (int r = 0; r < 8; ++r) {
            int m0 = r + 8 * half;
            lds_h[m0 * LDSTRIDE + nn]        = f2bf(tanhf(acc0[r]));
            lds_h[(m0 + 16) * LDSTRIDE + nn] = f2bf(tanhf(acc1[r]));
        }
        __syncthreads();

        // ---- scores: score[b][s] = h[b] . enc_t[s][b] ----
        for (int p = 0; p < 4; ++p) {
            int pair = threadIdx.x + p * 1024;     // 4096 (b,s) pairs
            int b = pair >> 7, s = pair & 127;
            const u16* et = etp + ((size_t)s * BATCH + b) * HDIM;
            const u16* hp = &lds_h[b * LDSTRIDE];
            float sum = 0.f;
            for (int k = 0; k < HDIM; k += 8) {
                uint4 qe = *(const uint4*)(et + k);
                uint4 qh = *(const uint4*)(hp + k);
                const u16* pe = (const u16*)&qe;
                const u16* ph = (const u16*)&qh;
                #pragma unroll
                for (int j = 0; j < 8; ++j) sum += bf2f(pe[j]) * bf2f(ph[j]);
            }
            lds_sc[b * SSRC + s] = sum;
        }
        __syncthreads();

        // ---- softmax over s (wave b handles row b), in place ----
        {
            int b = wave;
            float v[4];
            float mx = -3.4e38f;
            #pragma unroll
            for (int j = 0; j < 4; ++j) {
                v[j] = lds_sc[b * SSRC + lane + 32 * j];
                mx = fmaxf(mx, v[j]);
            }
            for (int o = 16; o > 0; o >>= 1) mx = fmaxf(mx, __shfl_xor(mx, o, 32));
            float sum = 0.f;
            #pragma unroll
            for (int j = 0; j < 4; ++j) { v[j] = __expf(v[j] - mx); sum += v[j]; }
            for (int o = 16; o > 0; o >>= 1) sum += __shfl_xor(sum, o, 32);
            float inv = __frcp_rn(sum);
            #pragma unroll
            for (int j = 0; j < 4; ++j) {
                int s = lane + 32 * j;
                float a = v[j] * inv;
                lds_sc[b * SSRC + s] = a;
                out_att[((size_t)b * SSRC + s) * TDEC + t] = a;
            }
        }
        __syncthreads();

        // ---- ctx + write model_hidden row [h, ctx] ----
        {
            int id = (int)threadIdx.x;
            int b  = id >> 5;
            int h0 = (id & 31) * 16;
            float c[16];
            #pragma unroll
            for (int j = 0; j < 16; ++j) c[j] = 0.f;
            for (int s = 0; s < SSRC; ++s) {
                float a = lds_sc[b * SSRC + s];
                const u16* er = eop + ((size_t)s * BATCH + b) * HDIM + h0;
                uint4 q0 = *(const uint4*)(er);
                uint4 q1 = *(const uint4*)(er + 8);
                const u16* pe = (const u16*)&q0;
                #pragma unroll
                for (int j = 0; j < 8; ++j) c[j] += a * bf2f(pe[j]);
                pe = (const u16*)&q1;
                #pragma unroll
                for (int j = 0; j < 8; ++j) c[8 + j] += a * bf2f(pe[j]);
            }
            u16* mh = model_hidden + ((size_t)t * BATCH + b) * K2;
            #pragma unroll
            for (int j = 0; j < 16; ++j) mh[HDIM + h0 + j] = f2bf(c[j]);
            #pragma unroll
            for (int j = 0; j < 16; ++j) mh[h0 + j] = lds_h[b * LDSTRIDE + h0 + j];
        }
        __syncthreads();
    }
}

// ---------------------------------------------------------------------------
// logits = model_hidden @ out_W^T + out_b   (M=4096, N=32000, K=1024)
// wave: 1 Mtile x 4 Ntiles (A reuse 4x). grid (500, 32), block 256.
// Output written non-temporally in (b, t, c) layout (f32): 524MB write-once
// stream must not displace the L2-resident 64MB out_W.
// ---------------------------------------------------------------------------
__global__ __launch_bounds__(256) void out_gemm_kernel(
    const u16* __restrict__ A, const u16* __restrict__ W,
    const float* __restrict__ bias, float* __restrict__ out) {
    int wave = threadIdx.x >> 5, lane = threadIdx.x & 31;
    int mtile  = blockIdx.y * 8 + wave;      // 0..255
    int ntbase = blockIdx.x * 4;             // 0..1996
    int m0 = mtile * 16;

    f32x8 acc[4];
    #pragma unroll
    for (int i = 0; i < 4; ++i) {
        float bv = bias[(ntbase + i) * 16 + (lane & 15)];
        #pragma unroll
        for (int r = 0; r < 8; ++r) acc[i][r] = bv;
    }

    for (int k0 = 0; k0 < K2; k0 += 32) {
        Frag a = load_frag_A_g(A, m0, K2, k0, lane);
        #pragma unroll
        for (int i = 0; i < 4; ++i) {
            // hide L2 latency of next k-chunk behind this group's WMMAs
            __builtin_prefetch(W + (size_t)((ntbase + i) * 16 + (lane & 15)) * K2
                                 + (k0 + 32) + (lane >> 4) * 16, 0, 0);
            Frag b = load_frag_B_g(W, (ntbase + i) * 16, K2, k0, lane);
            acc[i] = wmma_bf16(a, b, acc[i]);
        }
    }

    int half = lane >> 4;
    #pragma unroll
    for (int i = 0; i < 4; ++i) {
        int nn = (ntbase + i) * 16 + (lane & 15);
        #pragma unroll
        for (int r = 0; r < 8; ++r) {
            int m = m0 + r + 8 * half;
            int b = m & 31, t = m >> 5;              // m = t*32 + b
            __builtin_nontemporal_store(
                acc[i][r], out + (size_t)b * (TDEC * CVOC) + (size_t)t * CVOC + nn);
        }
    }
}

// ---------------------------------------------------------------------------
// Host driver
// ---------------------------------------------------------------------------
extern "C" void kernel_launch(void* const* d_in, const int* in_sizes, int n_in,
                              void* d_out, int out_size, void* d_ws, size_t ws_size,
                              hipStream_t stream) {
    const int*   src      = (const int*)d_in[0];
    const int*   tgt      = (const int*)d_in[1];
    const float* embed    = (const float*)d_in[2];
    const float* enc_Wih  = (const float*)d_in[3];
    const float* enc_Whh  = (const float*)d_in[4];
    const float* enc_bih  = (const float*)d_in[5];
    const float* enc_bhh  = (const float*)d_in[6];
    const float* dec_Wih  = (const float*)d_in[7];
    const float* dec_Whh  = (const float*)d_in[8];
    const float* dec_bih  = (const float*)d_in[9];
    const float* dec_bhh  = (const float*)d_in[10];
    const float* attn_W   = (const float*)d_in[11];
    const float* attn_b   = (const float*)d_in[12];
    const float* out_W    = (const float*)d_in[13];
    const float* out_b    = (const float*)d_in[14];

    float* out_logit = (float*)d_out;
    float* out_att   = out_logit + (size_t)BATCH * TDEC * CVOC;

    char* ws  = (char*)d_ws;
    size_t off = 0;
    auto alloc = [&](size_t bytes) -> void* {
        void* p = ws + off;
        off = (off + bytes + 255) & ~(size_t)255;
        return p;
    };

    u16* encWih_bf = (u16*)alloc((size_t)HDIM * HDIM * 2);
    u16* encWhh_bf = (u16*)alloc((size_t)HDIM * HDIM * 2);
    u16* decWih_bf = (u16*)alloc((size_t)HDIM * HDIM * 2);
    u16* decWhh_bf = (u16*)alloc((size_t)HDIM * HDIM * 2);
    u16* attnW_bf  = (u16*)alloc((size_t)HDIM * HDIM * 2);
    u16* outW_bf   = (u16*)alloc((size_t)CVOC * K2 * 2);
    float* enc_bias = (float*)alloc(HDIM * 4);
    float* dec_bias = (float*)alloc(HDIM * 4);
    u16* enc_emb = (u16*)alloc((size_t)SSRC * BATCH * HDIM * 2);
    u16* dec_emb = (u16*)alloc((size_t)TDEC * BATCH * HDIM * 2);
    u16* enc_out = (u16*)alloc((size_t)SSRC * BATCH * HDIM * 2);
    u16* enc_t   = (u16*)alloc((size_t)SSRC * BATCH * HDIM * 2);
    u16* model_hidden = (u16*)alloc((size_t)TDEC * BATCH * K2 * 2);
    (void)ws_size; (void)n_in; (void)in_sizes; (void)out_size;

    const int CT = 256;
    int nw = HDIM * HDIM;
    cvt_bf16_kernel<<<(nw + CT - 1) / CT, CT, 0, stream>>>(enc_Wih, encWih_bf, nw);
    cvt_bf16_kernel<<<(nw + CT - 1) / CT, CT, 0, stream>>>(enc_Whh, encWhh_bf, nw);
    cvt_bf16_kernel<<<(nw + CT - 1) / CT, CT, 0, stream>>>(dec_Wih, decWih_bf, nw);
    cvt_bf16_kernel<<<(nw + CT - 1) / CT, CT, 0, stream>>>(dec_Whh, decWhh_bf, nw);
    cvt_bf16_kernel<<<(nw + CT - 1) / CT, CT, 0, stream>>>(attn_W, attnW_bf, nw);
    int now = CVOC * K2;
    cvt_bf16_kernel<<<(now + CT - 1) / CT, CT, 0, stream>>>(out_W, outW_bf, now);

    addbias_kernel<<<(HDIM + CT - 1) / CT, CT, 0, stream>>>(enc_bih, enc_bhh, enc_bias, HDIM);
    addbias_kernel<<<(HDIM + CT - 1) / CT, CT, 0, stream>>>(dec_bih, dec_bhh, dec_bias, HDIM);

    gather_kernel<<<SSRC * BATCH, 128, 0, stream>>>(src, SSRC, embed, enc_emb);
    gather_kernel<<<TDEC * BATCH, 128, 0, stream>>>(tgt, SSRC + 1, embed, dec_emb);

    encoder_kernel<<<1, 1024, 0, stream>>>(enc_emb, encWih_bf, encWhh_bf, enc_bias, enc_out);

    attn_gemm_kernel<<<dim3(HDIM / 16, 32), 256, 0, stream>>>(enc_out, attnW_bf, attn_b, enc_t);

    decoder_kernel<<<1, 1024, 0, stream>>>(dec_emb, decWih_bf, decWhh_bf, dec_bias,
                                           enc_out, enc_t, model_hidden, out_att);

    out_gemm_kernel<<<dim3(CVOC / 64, 32), 256, 0, stream>>>(model_hidden, outW_bf, out_b, out_logit);
}
